// GPT_6150393168518
// MI455X (gfx1250) — compile-verified, matrix operand order checked by reading, not run on hardware
//
#include <hip/hip_runtime.h>
#include <hip/hip_bf16.h>
#include <math.h>

// ---------------------------------------------------------------------------
// Model constants (from reference)
// ---------------------------------------------------------------------------
#define GB 2
#define GT 1024
#define GV 50304
#define GC 768
#define GH 12
#define GL 4
#define GD 64
#define BT2 2048L   // B*T

typedef unsigned short u16t;
typedef __attribute__((ext_vector_type(16))) __bf16 v16bf;
typedef __attribute__((ext_vector_type(16))) u16t   v16u;
typedef __attribute__((ext_vector_type(8)))  u16t   v8u;
typedef __attribute__((ext_vector_type(8)))  float  v8f;

// fp32 -> bf16, round-to-nearest-even, inf/nan passthrough
__device__ __forceinline__ u16t f2bf(float f) {
  unsigned u = __builtin_bit_cast(unsigned, f);
  unsigned r = u + 0x7fffu + ((u >> 16) & 1u);
  if ((u & 0x7f800000u) == 0x7f800000u) r = u;
  return (u16t)(r >> 16);
}

// Build a 16-element bf16 fragment from one LDS row (ISA 16-bit A/B layout:
// vector elems 0..7 hold K = half*8 + j, elems 8..15 hold K = 16 + half*8 + j).
__device__ __forceinline__ v16bf ldfrag(const u16t* rowbase, int half) {
  v8u lo = *(const v8u*)(rowbase + half * 8);
  v8u hi = *(const v8u*)(rowbase + 16 + half * 8);
  v16u u;
#pragma unroll
  for (int j = 0; j < 8; ++j) { u[j] = lo[j]; u[j + 8] = hi[j]; }
  return __builtin_bit_cast(v16bf, u);
}

__device__ __forceinline__ v8f wmma_bf16(v16bf a, v16bf b, v8f c) {
  return __builtin_amdgcn_wmma_f32_16x16x32_bf16(false, a, false, b,
                                                 (short)0, c, false, false);
}

// ---------------------------------------------------------------------------
// Generic batched WMMA GEMM:
//   C[z,m,n] = scale * sum_k A[z,m,k]*B(z,k,n) + addsc[aidx] + bias[n]
//              + mask[z/mdiv, m, n]
// A: bf16 row-major [M,K] (lda), batch stride bA.
// B: bKN==0 -> weight W[N,K] (ldb = K-stride); bKN==1 -> row-major [K,N].
//
// Block: 128 threads = 4 waves. Block tile 128(M) x 64(N), K-stage 64.
// Each wave owns a 32x64 output: 2x4 WMMA tiles, 8 accumulators, and runs
// two 16x16x32 K-steps per stage -> 16 v_wmma per stage per wave, with A
// fragments reused across 4 B fragments. LDS is double-buffered, a single
// barrier per stage; the next stage's global loads issue before the WMMAs so
// VMEM latency hides under matrix ops. All K in this model are %64 == 0.
// ---------------------------------------------------------------------------
__global__ __launch_bounds__(128) void gemm_wmma(
    const u16t* __restrict__ A, long bA, int lda,
    const u16t* __restrict__ Bm, long bB, int ldb, int bKN,
    float* __restrict__ Cf, u16t* __restrict__ Cbf, long bC, int ldc,
    const float* __restrict__ bias,
    const float* __restrict__ addsc, int aidx,
    const float* __restrict__ mask, int mdiv, long mstride,
    float scale, int M, int N, int K)
{
  __shared__ __align__(16) u16t As[2][128][72];  // 64 + 8 pad (16B aligned rows)
  __shared__ __align__(16) u16t Ws[2][64][72];

  const int z    = blockIdx.z;
  const u16t* Ab = A  + (long)z * bA;
  const u16t* Bb = Bm + (long)z * bB;
  const int n0   = blockIdx.x * 64;
  const int m0   = blockIdx.y * 128;
  const int tid  = threadIdx.x;
  const int lane = tid & 31;
  const int wid  = tid >> 5;          // 4 waves
  const int wm   = wid << 5;          // 0,32,64,96
  const int half = lane >> 4;
  const int lr   = lane & 15;

  // ---- cooperative-load thread mapping (hoisted pointers) ----
  const int tr = tid >> 3;            // 0..15 (row group)
  const int tk = (tid & 7) * 8;       // 16B-vector offset within a 64-wide row
  const u16t* pa = Ab + (long)(m0 + tr) * lda + tk;
  const u16t* pw = bKN ? (Bb + (long)tr * ldb + (n0 + tk))
                       : (Bb + (long)(n0 + tr) * ldb + tk);
  const long wrstride = (long)16 * ldb;            // i-step for both layouts
  const long wstage   = bKN ? (long)64 * ldb : 64L;

  v8u rA[8], rW[4];
  auto stage_load = [&]() {
#pragma unroll
    for (int i = 0; i < 8; ++i) rA[i] = *(const v8u*)(pa + (long)16 * i * lda);
#pragma unroll
    for (int i = 0; i < 4; ++i) rW[i] = *(const v8u*)(pw + wrstride * i);
    pa += 64; pw += wstage;
  };
  auto stage_store = [&](int pp) {
#pragma unroll
    for (int i = 0; i < 8; ++i) *(v8u*)&As[pp][tr + 16 * i][tk] = rA[i];
    if (bKN) {            // transpose [K,N] operand into Ws[n][k] at store time
#pragma unroll
      for (int i = 0; i < 4; ++i)
#pragma unroll
        for (int j = 0; j < 8; ++j) Ws[pp][tk + j][tr + 16 * i] = rW[i][j];
    } else {
#pragma unroll
      for (int i = 0; i < 4; ++i) *(v8u*)&Ws[pp][tr + 16 * i][tk] = rW[i];
    }
  };

  v8f c[2][4];
#pragma unroll
  for (int i = 0; i < 2; ++i)
#pragma unroll
    for (int j = 0; j < 4; ++j) c[i][j] = (v8f){0.f,0.f,0.f,0.f,0.f,0.f,0.f,0.f};

  // ---- prologue: stage k0=0 into buffer 0 ----
  stage_load();
  stage_store(0);
  __syncthreads();

  int p = 0;
  for (int k0 = 0; k0 < K; k0 += 64) {
    const bool hasNext = (k0 + 64) < K;
    if (hasNext) stage_load();          // issue next stage's global loads early

#pragma unroll
    for (int kk = 0; kk < 64; kk += 32) {
      v16bf a0 = ldfrag(&As[p][wm + lr][kk],      half);
      v16bf a1 = ldfrag(&As[p][wm + 16 + lr][kk], half);
      v16bf b0 = ldfrag(&Ws[p][lr][kk],           half);
      v16bf b1 = ldfrag(&Ws[p][16 + lr][kk],      half);
      v16bf b2 = ldfrag(&Ws[p][32 + lr][kk],      half);
      v16bf b3 = ldfrag(&Ws[p][48 + lr][kk],      half);
      c[0][0] = wmma_bf16(a0, b0, c[0][0]);
      c[0][1] = wmma_bf16(a0, b1, c[0][1]);
      c[0][2] = wmma_bf16(a0, b2, c[0][2]);
      c[0][3] = wmma_bf16(a0, b3, c[0][3]);
      c[1][0] = wmma_bf16(a1, b0, c[1][0]);
      c[1][1] = wmma_bf16(a1, b1, c[1][1]);
      c[1][2] = wmma_bf16(a1, b2, c[1][2]);
      c[1][3] = wmma_bf16(a1, b3, c[1][3]);
    }

    if (hasNext) stage_store(p ^ 1);    // drain staged registers to other buffer
    __syncthreads();
    p ^= 1;
  }

  // ---- fused epilogue (C layout: lane n = lr, VGPR r -> m = half*8 + r) ----
  const float addc  = (addsc != nullptr) ? addsc[aidx] : 0.f;
  const float* mrow = (mask != nullptr) ? (mask + (long)(z / mdiv) * mstride) : nullptr;
#pragma unroll
  for (int i = 0; i < 2; ++i) {
#pragma unroll
    for (int j = 0; j < 4; ++j) {
      const int n = n0 + j * 16 + lr;
      float bvv = (bias != nullptr) ? bias[n] : 0.f;
#pragma unroll
      for (int r = 0; r < 8; ++r) {
        int m   = m0 + wm + i * 16 + half * 8 + r;
        float v = c[i][j][r] * scale + addc + bvv;
        if (mrow) v += mrow[(long)m * N + n];
        long off = (long)z * bC + (long)m * ldc + n;
        if (Cf)  Cf[off]  = v;
        if (Cbf) Cbf[off] = f2bf(v);
      }
    }
  }
}

// ---------------------------------------------------------------------------
// Embedding: x[b,t,c] = wte[idx[b,t],c] + wpe[t,c]
// ---------------------------------------------------------------------------
__global__ void embed_k(const float* __restrict__ wte, const float* __restrict__ wpe,
                        const int* __restrict__ idx, float* __restrict__ x)
{
  long i = (long)blockIdx.x * 256 + threadIdx.x;
  if (i >= BT2 * GC) return;
  long row = i / GC; int c = (int)(i % GC);
  int t = (int)(row % GT);
  x[i] = wte[(long)idx[row] * GC + c] + wpe[(long)t * GC + c];
}

// fp32 -> bf16 bulk convert
__global__ void cvt_bf(const float* __restrict__ s, u16t* __restrict__ d, long n)
{
  long i = (long)blockIdx.x * 256 + threadIdx.x;
  if (i < n) d[i] = f2bf(s[i]);
}

// ---------------------------------------------------------------------------
// LayerNorm row kernel (one block per row), bf16 output (matmul operand only)
// ---------------------------------------------------------------------------
__global__ __launch_bounds__(256) void layernorm_bf(
    const float* __restrict__ x, const float* __restrict__ w,
    const float* __restrict__ b, u16t* __restrict__ o, int Cdim)
{
  __shared__ float red[256];
  long row = blockIdx.x;
  const float* xr = x + row * (long)Cdim;
  int t = threadIdx.x;
  float s = 0.f;
  for (int c = t; c < Cdim; c += 256) s += xr[c];
  red[t] = s; __syncthreads();
  for (int st = 128; st > 0; st >>= 1) { if (t < st) red[t] += red[t + st]; __syncthreads(); }
  float mu = red[0] / (float)Cdim; __syncthreads();
  float s2 = 0.f;
  for (int c = t; c < Cdim; c += 256) { float d = xr[c] - mu; s2 += d * d; }
  red[t] = s2; __syncthreads();
  for (int st = 128; st > 0; st >>= 1) { if (t < st) red[t] += red[t + st]; __syncthreads(); }
  float inv = rsqrtf(red[0] / (float)Cdim + 1e-5f);
  for (int c = t; c < Cdim; c += 256)
    o[row * (long)Cdim + c] = f2bf((xr[c] - mu) * inv * w[c] + b[c]);
}

// split qkv -> per-head bf16 [B,H,T,D]
__global__ void split_qkv(const float* __restrict__ qkv, u16t* __restrict__ qb,
                          u16t* __restrict__ kb, u16t* __restrict__ vb)
{
  long i = (long)blockIdx.x * 256 + threadIdx.x;
  if (i >= BT2 * GC) return;
  long row = i / GC; int c = (int)(i % GC);
  int h = c / GD, d = c % GD;
  long b = row / GT, t = row % GT;
  long dst = (((b * GH + h) * GT + t) * GD + d);
  const float* src = qkv + row * (3L * GC);
  qb[dst] = f2bf(src[c]);
  kb[dst] = f2bf(src[GC + c]);
  vb[dst] = f2bf(src[2 * GC + c]);
}

// merge heads [B,H,T,D] fp32 -> [B*T, C] bf16
__global__ void merge_heads(const float* __restrict__ ybh, u16t* __restrict__ yb)
{
  long i = (long)blockIdx.x * 256 + threadIdx.x;
  if (i >= BT2 * GC) return;
  long row = i / GC; int c = (int)(i % GC);
  int h = c / GD, d = c % GD;
  long b = row / GT, t = row % GT;
  yb[i] = f2bf(ybh[((b * GH + h) * GT + t) * GD + d]);
}

// ---------------------------------------------------------------------------
// 2-way alpha-entmax via bisection on (raw, 0); strict-lower gate (p=1 at k>=q)
// In-place on raw buffer [B,T,T].
// ---------------------------------------------------------------------------
__global__ void entmax2_k(float* __restrict__ raw)
{
  long i = (long)blockIdx.x * 256 + threadIdx.x;
  if (i >= (long)GB * GT * GT) return;
  int kcol = (int)(i % GT);
  int q    = (int)((i / GT) % GT);
  float p;
  if (kcol >= q) {
    p = 1.0f;
  } else {
    const float am1 = (float)(1.000001 - 1.0);
    const float inv = 1.0f / am1;
    float Xs0 = raw[i] * am1;                       // Xs1 = 0
    float mx  = fmaxf(Xs0, 0.f);
    float tlo = mx - 1.f;
    float thi = mx - powf(0.5f, am1);
    float flo = powf(fmaxf(Xs0 - tlo, 0.f), inv) + powf(fmaxf(-tlo, 0.f), inv) - 1.f;
    float dm  = thi - tlo;
    float p0 = 0.f, p1 = 0.f;
    for (int it = 0; it < 50; ++it) {
      dm *= 0.5f;
      float tm = tlo + dm;
      p0 = powf(fmaxf(Xs0 - tm, 0.f), inv);
      p1 = powf(fmaxf(-tm, 0.f), inv);
      float fm = p0 + p1 - 1.f;
      if (fm * flo >= 0.f) tlo = tm;
    }
    p = p0 / (p0 + p1);
  }
  raw[i] = p;
}

// cumprod over queries + log + depth-propagated mask + causal -inf (in place)
__global__ void cummask_k(const float* __restrict__ p, float* __restrict__ mask, int first)
{
  int col = blockIdx.x * 256 + threadIdx.x;         // over B*T columns
  if (col >= GB * GT) return;
  int b = col / GT, kcol = col % GT;
  const float* pb = p + (long)b * GT * GT;
  float* mb = mask + (long)b * GT * GT;
  float c = 1.f;
  for (int q = 0; q < GT; ++q) {
    c *= pb[(long)q * GT + kcol];
    float v = logf(c + 1e-40f);
    if (!first) v += mb[(long)q * GT + kcol];
    mb[(long)q * GT + kcol] = (kcol <= q) ? v : -__builtin_inff();
  }
}

// row softmax [*, T] fp32 -> bf16
__global__ __launch_bounds__(256) void softmax_bf16(
    const float* __restrict__ S, u16t* __restrict__ A, int T)
{
  __shared__ float red[256];
  long row = blockIdx.x;
  const float* sr = S + row * (long)T;
  int t = threadIdx.x;
  float mx = -3.4e38f;
  for (int c = t; c < T; c += 256) mx = fmaxf(mx, sr[c]);
  red[t] = mx; __syncthreads();
  for (int st = 128; st > 0; st >>= 1) { if (t < st) red[t] = fmaxf(red[t], red[t + st]); __syncthreads(); }
  float m = red[0]; __syncthreads();
  float sum = 0.f;
  for (int c = t; c < T; c += 256) sum += __expf(sr[c] - m);
  red[t] = sum; __syncthreads();
  for (int st = 128; st > 0; st >>= 1) { if (t < st) red[t] += red[t + st]; __syncthreads(); }
  float inv = 1.0f / red[0];
  for (int c = t; c < T; c += 256)
    A[row * (long)T + c] = f2bf(__expf(sr[c] - m) * inv);
}

// tanh-GELU fp32 -> bf16
__global__ void gelu_bf(const float* __restrict__ x, u16t* __restrict__ o, long n)
{
  long i = (long)blockIdx.x * 256 + threadIdx.x;
  if (i >= n) return;
  float v = x[i];
  float g = 0.5f * v * (1.f + tanhf(0.7978845608028654f * (v + 0.044715f * v * v * v)));
  o[i] = f2bf(g);
}

// residual add
__global__ void add_k(float* __restrict__ x, const float* __restrict__ y, long n)
{
  long i = (long)blockIdx.x * 256 + threadIdx.x;
  if (i < n) x[i] += y[i];
}

// ---------------------------------------------------------------------------
// Host-side orchestration
// ---------------------------------------------------------------------------
static inline void launch_gemm(hipStream_t s,
    const u16t* A, long bA, int lda,
    const u16t* Bm, long bB, int ldb, int bKN,
    float* Cf, u16t* Cbf, long bC, int ldc,
    const float* bias, const float* addsc, int aidx,
    const float* mask, int mdiv, long mstride,
    float scale, int M, int N, int K, int batch)
{
  dim3 g((unsigned)(N / 64), (unsigned)(M / 128), (unsigned)batch);
  gemm_wmma<<<g, dim3(128), 0, s>>>(A, bA, lda, Bm, bB, ldb, bKN,
                                    Cf, Cbf, bC, ldc, bias, addsc, aidx,
                                    mask, mdiv, mstride, scale, M, N, K);
}

static inline unsigned gblk(long n) { return (unsigned)((n + 255) / 256); }

extern "C" void kernel_launch(void* const* d_in, const int* in_sizes, int n_in,
                              void* d_out, int out_size, void* d_ws, size_t ws_size,
                              hipStream_t stream)
{
  (void)in_sizes; (void)n_in; (void)out_size; (void)ws_size;
  const float* wte   = (const float*)d_in[0];
  const float* wpe   = (const float*)d_in[1];
  const float* ln1w  = (const float*)d_in[2];
  const float* ln1b  = (const float*)d_in[3];
  const float* attnw = (const float*)d_in[4];
  const float* attnb = (const float*)d_in[5];
  const float* qintw = (const float*)d_in[6];
  const float* kintw = (const float*)d_in[7];
  const float* intb  = (const float*)d_in[8];
  const float* projw = (const float*)d_in[9];
  const float* projb = (const float*)d_in[10];
  const float* ln2w  = (const float*)d_in[11];
  const float* ln2b  = (const float*)d_in[12];
  const float* fcw   = (const float*)d_in[13];
  const float* fcb   = (const float*)d_in[14];
  const float* fc2w  = (const float*)d_in[15];
  const float* fc2b  = (const float*)d_in[16];
  const float* lnfw  = (const float*)d_in[17];
  const float* lnfb  = (const float*)d_in[18];
  const int*   idx   = (const int*)d_in[19];
  float* out = (float*)d_out;

  // ---- workspace carve-out ----
  char* wp = (char*)d_ws;
  auto alloc = [&](size_t bytes) -> char* {
    char* r = wp; wp += (bytes + 255) & ~(size_t)255; return r;
  };
  u16t* wte_bf   = (u16t*)alloc((size_t)GV * GC * 2);
  u16t* attnw_bf = (u16t*)alloc((size_t)GL * 3 * GC * GC * 2);
  u16t* qintw_bf = (u16t*)alloc((size_t)GL * GD * GC * 2);
  u16t* kintw_bf = (u16t*)alloc((size_t)GL * GD * GC * 2);
  u16t* projw_bf = (u16t*)alloc((size_t)GL * GC * GC * 2);
  u16t* fcw_bf   = (u16t*)alloc((size_t)GL * 4 * GC * GC * 2);
  u16t* fc2w_bf  = (u16t*)alloc((size_t)GL * 4 * GC * GC * 2);
  float* x       = (float*)alloc((size_t)BT2 * GC * 4);
  u16t* h_bf     = (u16t*)alloc((size_t)BT2 * GC * 2);
  float* qkv     = (float*)alloc((size_t)BT2 * 3 * GC * 4);   // reused as fc2-out scratch
  u16t* q_bf     = (u16t*)alloc((size_t)BT2 * GC * 2);
  u16t* k_bf     = (u16t*)alloc((size_t)BT2 * GC * 2);
  u16t* v_bf     = (u16t*)alloc((size_t)BT2 * GC * 2);
  u16t* qint_bf  = (u16t*)alloc((size_t)BT2 * GD * 2);
  u16t* kint_bf  = (u16t*)alloc((size_t)BT2 * GD * 2);
  float* rawp    = (float*)alloc((size_t)GB * GT * GT * 4);
  float* maskb   = (float*)alloc((size_t)GB * GT * GT * 4);   // persists across layers
  float* scores  = (float*)alloc((size_t)GB * GH * GT * GT * 4);
  u16t* att_bf   = (u16t*)alloc((size_t)GB * GH * GT * GT * 2);
  float* ybh     = (float*)alloc((size_t)BT2 * GC * 4);
  u16t* y_bf     = (u16t*)alloc((size_t)BT2 * GC * 2);
  float* tmp     = (float*)alloc((size_t)BT2 * 4 * GC * 4);
  u16t* act_bf   = (u16t*)alloc((size_t)BT2 * 4 * GC * 2);

  // ---- one-time weight conversion to bf16 ----
  auto cvt = [&](const float* s, u16t* d, long n) {
    cvt_bf<<<gblk(n), 256, 0, stream>>>(s, d, n);
  };
  cvt(wte,   wte_bf,   (long)GV * GC);
  cvt(attnw, attnw_bf, (long)GL * 3 * GC * GC);
  cvt(qintw, qintw_bf, (long)GL * GD * GC);
  cvt(kintw, kintw_bf, (long)GL * GD * GC);
  cvt(projw, projw_bf, (long)GL * GC * GC);
  cvt(fcw,   fcw_bf,   (long)GL * 4 * GC * GC);
  cvt(fc2w,  fc2w_bf,  (long)GL * 4 * GC * GC);

  // ---- embedding ----
  embed_k<<<gblk(BT2 * GC), 256, 0, stream>>>(wte, wpe, idx, x);

  const float iscale = 0.125f;   // 1/sqrt(64) for both gate and attention scores
  const long TT = (long)GT * GT;

  for (int l = 0; l < GL; ++l) {
    // h = LN1(x)  (bf16 matmul operand)
    layernorm_bf<<<(unsigned)BT2, 256, 0, stream>>>(x, ln1w + l * GC, ln1b + l * GC, h_bf, GC);

    // qkv = h @ attn_w.T + attn_b
    launch_gemm(stream, h_bf, 0, GC,
                attnw_bf + (size_t)l * 3 * GC * GC, 0, GC, 0,
                qkv, nullptr, 0, 3 * GC,
                attnb + l * 3 * GC, nullptr, 0, nullptr, 1, 0,
                1.0f, (int)BT2, 3 * GC, GC, 1);
    split_qkv<<<gblk(BT2 * GC), 256, 0, stream>>>(qkv, q_bf, k_bf, v_bf);

    // integer-gate projections (no bias), bf16 out
    launch_gemm(stream, h_bf, 0, GC, qintw_bf + (size_t)l * GD * GC, 0, GC, 0,
                nullptr, qint_bf, 0, GD, nullptr, nullptr, 0, nullptr, 1, 0,
                1.0f, (int)BT2, GD, GC, 1);
    launch_gemm(stream, h_bf, 0, GC, kintw_bf + (size_t)l * GD * GC, 0, GC, 0,
                nullptr, kint_bf, 0, GD, nullptr, nullptr, 0, nullptr, 1, 0,
                1.0f, (int)BT2, GD, GC, 1);

    // raw = qint @ kint.T / sqrt(DINT) + int_bias[l]   (batched over B)
    launch_gemm(stream, qint_bf, (long)GT * GD, GD,
                kint_bf, (long)GT * GD, GD, 0,
                rawp, nullptr, TT, GT,
                nullptr, intb, l, nullptr, 1, 0,
                iscale, GT, GT, GD, GB);

    // entmax gate + cumprod log-mask (with depth propagation)
    entmax2_k<<<gblk((long)GB * TT), 256, 0, stream>>>(rawp);
    cummask_k<<<gblk(GB * GT), 256, 0, stream>>>(rawp, maskb, l == 0 ? 1 : 0);

    // scores = q @ k.T / sqrt(hd) + mask   (batched over B*H; mask shared per B)
    launch_gemm(stream, q_bf, (long)GT * GD, GD,
                k_bf, (long)GT * GD, GD, 0,
                scores, nullptr, TT, GT,
                nullptr, nullptr, 0, maskb, GH, TT,
                iscale, GT, GT, GD, GB * GH);

    // att = softmax(scores), bf16
    softmax_bf16<<<(unsigned)(GB * GH * GT), 256, 0, stream>>>(scores, att_bf, GT);

    // y = att @ v   (B operand is [K=T, N=64] row-major per batch)
    launch_gemm(stream, att_bf, TT, GT,
                v_bf, (long)GT * GD, GD, 1,
                ybh, nullptr, (long)GT * GD, GD,
                nullptr, nullptr, 0, nullptr, 1, 0,
                1.0f, GT, GD, GT, GB * GH);
    merge_heads<<<gblk(BT2 * GC), 256, 0, stream>>>(ybh, y_bf);

    // x += y @ proj_w.T + proj_b
    launch_gemm(stream, y_bf, 0, GC, projw_bf + (size_t)l * GC * GC, 0, GC, 0,
                tmp, nullptr, 0, GC, projb + l * GC, nullptr, 0, nullptr, 1, 0,
                1.0f, (int)BT2, GC, GC, 1);
    add_k<<<gblk(BT2 * GC), 256, 0, stream>>>(x, tmp, BT2 * GC);

    // MLP: x += fc2(gelu(fc(LN2(x))))
    layernorm_bf<<<(unsigned)BT2, 256, 0, stream>>>(x, ln2w + l * GC, ln2b + l * GC, h_bf, GC);
    launch_gemm(stream, h_bf, 0, GC, fcw_bf + (size_t)l * 4 * GC * GC, 0, GC, 0,
                tmp, nullptr, 0, 4 * GC, fcb + l * 4 * GC, nullptr, 0, nullptr, 1, 0,
                1.0f, (int)BT2, 4 * GC, GC, 1);
    gelu_bf<<<gblk(BT2 * 4 * GC), 256, 0, stream>>>(tmp, act_bf, BT2 * 4 * GC);
    launch_gemm(stream, act_bf, 0, 4 * GC, fc2w_bf + (size_t)l * 4 * GC * GC, 0, 4 * GC, 0,
                qkv, nullptr, 0, GC, fc2b + l * GC, nullptr, 0, nullptr, 1, 0,
                1.0f, (int)BT2, GC, 4 * GC, 1);
    add_k<<<gblk(BT2 * GC), 256, 0, stream>>>(x, qkv, BT2 * GC);
  }

  // final LN + weight-tied lm_head: out = LN(x) @ wte.T
  layernorm_bf<<<(unsigned)BT2, 256, 0, stream>>>(x, lnfw, lnfb, h_bf, GC);
  launch_gemm(stream, h_bf, 0, GC, wte_bf, 0, GC, 0,
              out, nullptr, 0, GV, nullptr, nullptr, 0, nullptr, 1, 0,
              1.0f, (int)BT2, GV, GC, 1);
}
